// GATModel_47562467836509
// MI455X (gfx1250) — matrix operand order repaired and checked by reading, not compiled, not run on hardware
//
#include <hip/hip_runtime.h>
#include <hip/hip_bf16.h>

// ---------------- problem constants (match reference) ----------------
#define NN     50000
#define EE     800000
#define F_IN   128
#define HID    64
#define HEADS  4
#define HC     (HEADS*HID)   // 256
#define OUTC   16
#define BN_EPS 1e-5f
#define SM_EPS 1e-16f

typedef __attribute__((ext_vector_type(16))) __bf16 v16bf;
typedef __attribute__((ext_vector_type(8)))  float  v8f;
typedef __attribute__((ext_vector_type(4)))  unsigned int uint4v;

static __device__ __forceinline__ float lrelu(float x) { return x > 0.f ? x : 0.2f * x; }

static __device__ __forceinline__ unsigned short f2bf(float f) {
    unsigned int u = __float_as_uint(f);
    unsigned int r = u + 0x7FFFu + ((u >> 16) & 1u);   // round-to-nearest-even
    if ((u & 0x7F800000u) == 0x7F800000u) r = u;       // keep Inf/NaN bits
    return (unsigned short)(r >> 16);
}

static __device__ __forceinline__ void atomicMaxF(float* addr, float val) {
    unsigned int* ia = (unsigned int*)addr;
    unsigned int old = *ia;
    while (__uint_as_float(old) < val) {
        unsigned int assumed = old;
        old = atomicCAS(ia, assumed, __float_as_uint(val));
        if (old == assumed) break;
    }
}

// ---------------- elementwise helpers ----------------
// packed pair conversion: 2 floats -> 1 dword of bf16
__global__ __launch_bounds__(256) void k_cvt_bf16x2(const float* __restrict__ in,
                                                    unsigned int* __restrict__ out, int npairs) {
    int i = blockIdx.x * blockDim.x + threadIdx.x;
    if (i >= npairs) return;
    float2 f = *(const float2*)(in + 2 * i);
    out[i] = (unsigned int)f2bf(f.x) | ((unsigned int)f2bf(f.y) << 16);
}

__global__ __launch_bounds__(256) void k_zero(float* __restrict__ p, int n) {
    int i = blockIdx.x * blockDim.x + threadIdx.x;
    if (i < n) p[i] = 0.f;
}

// ---------------- bf16 WMMA GEMM: C[M,NCOLS] = A[M,KK] * B[KK,NCOLS] ----------------
// One wave -> one 16x16 tile; B slab staged transposed in LDS.
// All fragments loaded first (single clause), then the WMMA chain, so load
// latency overlaps the matrix ops. NCOLS compile-time -> immediate-offset stores.
template <int KK, int COLS, int NWAVES, int NCOLS>
__global__ __launch_bounds__(NWAVES * 32) void k_gemm_bf16_wmma(
        const unsigned short* __restrict__ A,
        const unsigned short* __restrict__ B,
        float* __restrict__ C) {
    constexpr int KP = KK + 8;                  // pad keeps columns 16B aligned, banks spread
    constexpr int NSTEP = KK / 32;
    __shared__ unsigned short Bt[COLS][KP];     // transposed: Bt[col][k]

    const int tid  = threadIdx.x;
    const int lane = tid & 31;
    const int wave = tid >> 5;
    const int row0 = blockIdx.x * 16;
    const int colBase = blockIdx.y * COLS;

    // cooperative, coalesced load of B slab; store transposed into LDS
    for (int idx = tid; idx < KK * COLS; idx += NWAVES * 32) {
        int k = idx / COLS, c = idx - k * COLS;
        Bt[c][k] = B[(size_t)k * NCOLS + colBase + c];
    }
    __syncthreads();

    const int l15 = lane & 15, sel = lane >> 4;
    const unsigned short* Arow = A + (size_t)(row0 + l15) * KK;  // lane = row M
    const unsigned short* Bcol = &Bt[wave * 16 + l15][0];        // lane = col N

    // phase 1: issue every fragment load (global b128 x2 + ds b128 x2 per step)
    uint4v ar[NSTEP][2], br[NSTEP][2];
#pragma unroll
    for (int s = 0; s < NSTEP; ++s) {
        const int ka = s * 32 + sel * 8;     // A: K = ka+0..7 and ka+16..23
        ar[s][0] = *(const uint4v*)(Arow + ka);
        ar[s][1] = *(const uint4v*)(Arow + ka + 16);
        const int kb = s * 32 + sel * 8;     // B: K = kb+0..7 and kb+16..23
        br[s][0] = *(const uint4v*)(Bcol + kb);
        br[s][1] = *(const uint4v*)(Bcol + kb + 16);
    }

    // phase 2: WMMA chain
    v8f acc = {};
#pragma unroll
    for (int s = 0; s < NSTEP; ++s) {
        union { uint4v u[2]; v16bf v; } av, bv;
        av.u[0] = ar[s][0]; av.u[1] = ar[s][1];
        bv.u[0] = br[s][0]; bv.u[1] = br[s][1];
        acc = __builtin_amdgcn_wmma_f32_16x16x32_bf16(false, av.v, false, bv.v,
                                                      (short)0, acc, false, false);
    }

    // stores: one base address, compile-time row-stride immediate offsets
    float* Crow = C + (size_t)(row0 + 8 * sel) * NCOLS + colBase + wave * 16 + l15;
#pragma unroll
    for (int v = 0; v < 8; ++v)
        Crow[v * NCOLS] = acc[v];
}

// ---------------- attention dot products: als/ald [N*H] ----------------
__global__ __launch_bounds__(256) void k_attn_dots(const float* __restrict__ h,
                                                   const float* __restrict__ a_s,
                                                   const float* __restrict__ a_d,
                                                   float* __restrict__ als,
                                                   float* __restrict__ ald,
                                                   int n_total, int H, int C) {
    int i = blockIdx.x * blockDim.x + threadIdx.x;
    if (i >= n_total) return;
    int hh = i % H;
    const float4* hp = (const float4*)(h + (size_t)i * C);   // i == n*H + hh
    const float4* as = (const float4*)(a_s + hh * C);
    const float4* ad = (const float4*)(a_d + hh * C);
    float s0 = 0.f, s1 = 0.f;
    for (int c = 0; c < C / 4; ++c) {
        float4 v = hp[c], w0 = as[c], w1 = ad[c];
        s0 += v.x * w0.x + v.y * w0.y + v.z * w0.z + v.w * w0.w;
        s1 += v.x * w1.x + v.y * w1.y + v.z * w1.z + v.w * w1.w;
    }
    als[i] = s0; ald[i] = s1;
}

// m init with the self-loop edge value (self-loop always present)
__global__ __launch_bounds__(256) void k_init_m(const float* __restrict__ als,
                                                const float* __restrict__ ald,
                                                float* __restrict__ m, int n) {
    int i = blockIdx.x * blockDim.x + threadIdx.x;
    if (i < n) m[i] = lrelu(als[i] + ald[i]);
}

// pass 1: e = leaky(als[src]+ald[dst]); store e; segment max by dst
__global__ __launch_bounds__(256) void k_edge_max(const int* __restrict__ ei,
                                                  const float* __restrict__ als,
                                                  const float* __restrict__ ald,
                                                  float* __restrict__ e_ws,
                                                  float* __restrict__ m,
                                                  int E, int H) {
    int i = blockIdx.x * blockDim.x + threadIdx.x;
    if (i >= E * H) return;
    int e = i / H, hh = i - e * H;
    int s = ei[e], d = ei[E + e];
    float v = lrelu(als[s * H + hh] + ald[d * H + hh]);
    e_ws[i] = v;
    atomicMaxF(&m[d * H + hh], v);
}

// denom init with the self-loop term
__global__ __launch_bounds__(256) void k_init_denom(const float* __restrict__ als,
                                                    const float* __restrict__ ald,
                                                    const float* __restrict__ m,
                                                    float* __restrict__ denom, int n) {
    int i = blockIdx.x * blockDim.x + threadIdx.x;
    if (i < n) denom[i] = __expf(lrelu(als[i] + ald[i]) - m[i]);
}

// pass 2: ex = exp(e - m[dst]); store; segment sum by dst
__global__ __launch_bounds__(256) void k_edge_expsum(const int* __restrict__ ei,
                                                     float* __restrict__ e_ws,
                                                     const float* __restrict__ m,
                                                     float* __restrict__ denom,
                                                     int E, int H) {
    int i = blockIdx.x * blockDim.x + threadIdx.x;
    if (i >= E * H) return;
    int e = i / H, hh = i - e * H;
    int d = ei[E + e];
    float ex = __expf(e_ws[i] - m[d * H + hh]);
    e_ws[i] = ex;
    atomicAdd(&denom[d * H + hh], ex);
}

// agg init = self-loop message (alpha_self * h[n]); float4 per thread
__global__ __launch_bounds__(256) void k_agg_init(const float* __restrict__ h,
                                                  const float* __restrict__ als,
                                                  const float* __restrict__ ald,
                                                  const float* __restrict__ m,
                                                  const float* __restrict__ denom,
                                                  float* __restrict__ agg,
                                                  int total4, int C) {
    int i = blockIdx.x * blockDim.x + threadIdx.x;
    if (i >= total4) return;
    int base = i * 4;
    int nh = base / C;
    float self = lrelu(als[nh] + ald[nh]);
    float alpha = __expf(self - m[nh]) / (denom[nh] + SM_EPS);
    float4 v = *(const float4*)(h + base);
    float4 o; o.x = alpha * v.x; o.y = alpha * v.y; o.z = alpha * v.z; o.w = alpha * v.w;
    *(float4*)(agg + base) = o;
}

// pass 3: agg[dst] += alpha * h[src]; float4 per thread
__global__ __launch_bounds__(256) void k_edge_aggregate(const int* __restrict__ ei,
                                                        const float* __restrict__ e_ws,
                                                        const float* __restrict__ denom,
                                                        const float* __restrict__ h,
                                                        float* __restrict__ agg,
                                                        int E, int H, int C) {
    const int HCl = H * C, HCl4 = HCl / 4;
    int i = blockIdx.x * blockDim.x + threadIdx.x;
    if (i >= E * HCl4) return;
    int e = i / HCl4, q = i - e * HCl4;          // q: float4 index within row
    int hh = (q * 4) / C;
    int s = ei[e], d = ei[E + e];
    float alpha = e_ws[e * H + hh] / (denom[d * H + hh] + SM_EPS);
    float4 v = *(const float4*)(h + (size_t)s * HCl + q * 4);
    float* dst = agg + (size_t)d * HCl + q * 4;
    atomicAdd(dst + 0, alpha * v.x);
    atomicAdd(dst + 1, alpha * v.y);
    atomicAdd(dst + 2, alpha * v.z);
    atomicAdd(dst + 3, alpha * v.w);
}

// ---------------- batchnorm ----------------
__global__ __launch_bounds__(256) void k_bn_stats(const float* __restrict__ agg,
                                                  const float* __restrict__ b1,
                                                  float* __restrict__ sum,
                                                  float* __restrict__ sumsq, int N) {
    int c = threadIdx.x;                          // 256 channels
    int rows = (N + gridDim.x - 1) / gridDim.x;
    int r0 = blockIdx.x * rows;
    int r1 = r0 + rows; if (r1 > N) r1 = N;
    float bb = b1[c], s = 0.f, s2 = 0.f;
    for (int r = r0; r < r1; ++r) {
        float v = agg[(size_t)r * HC + c] + bb;
        s += v; s2 += v * v;
    }
    atomicAdd(&sum[c], s); atomicAdd(&sumsq[c], s2);
}

__global__ __launch_bounds__(256) void k_bn_final(const float* __restrict__ sum,
                                                  const float* __restrict__ sumsq,
                                                  const float* __restrict__ gamma,
                                                  const float* __restrict__ beta,
                                                  float* __restrict__ scale,
                                                  float* __restrict__ shift) {
    int c = threadIdx.x;
    float mu = sum[c] / (float)NN;
    float var = sumsq[c] / (float)NN - mu * mu;
    float inv = rsqrtf(var + BN_EPS);
    float sc = gamma[c] * inv;
    scale[c] = sc;
    shift[c] = beta[c] - mu * sc;
}

// normalize + ReLU + bf16 convert, 2 channels per thread (packed dword store)
__global__ __launch_bounds__(256) void k_norm_relu_bf16(const float* __restrict__ agg,
                                                        const float* __restrict__ b1,
                                                        const float* __restrict__ scale,
                                                        const float* __restrict__ shift,
                                                        unsigned int* __restrict__ hn, int npairs) {
    int i = blockIdx.x * blockDim.x + threadIdx.x;
    if (i >= npairs) return;
    int c = (2 * i) & (HC - 1);
    float2 v = *(const float2*)(agg + 2 * i);
    float a = fmaxf((v.x + b1[c])     * scale[c]     + shift[c],     0.f);
    float b = fmaxf((v.y + b1[c + 1]) * scale[c + 1] + shift[c + 1], 0.f);
    hn[i] = (unsigned int)f2bf(a) | ((unsigned int)f2bf(b) << 16);
}

__global__ __launch_bounds__(256) void k_bias_out(float* __restrict__ out,
                                                  const float* __restrict__ b2, int total) {
    int i = blockIdx.x * blockDim.x + threadIdx.x;
    if (i < total) out[i] += b2[i & (OUTC - 1)];
}

// ---------------- launcher ----------------
extern "C" void kernel_launch(void* const* d_in, const int* in_sizes, int n_in,
                              void* d_out, int out_size, void* d_ws, size_t ws_size,
                              hipStream_t stream) {
    const float* x      = (const float*)d_in[0];
    const int*   ei     = (const int*)d_in[1];    // [2,E]
    const float* W1     = (const float*)d_in[2];
    const float* a_src1 = (const float*)d_in[3];
    const float* a_dst1 = (const float*)d_in[4];
    const float* b1     = (const float*)d_in[5];
    const float* gamma  = (const float*)d_in[6];
    const float* beta   = (const float*)d_in[7];
    const float* W2     = (const float*)d_in[8];
    const float* a_src2 = (const float*)d_in[9];
    const float* a_dst2 = (const float*)d_in[10];
    const float* b2     = (const float*)d_in[11];
    float* out = (float*)d_out;

    // ---- workspace carve-up (256B aligned) ----
    char* w = (char*)d_ws; size_t off = 0;
    auto alloc = [&](size_t bytes) -> void* {
        void* p = w + off; off += (bytes + 255) & ~(size_t)255; return p;
    };
    unsigned short* xb   = (unsigned short*)alloc((size_t)NN * F_IN * 2);
    unsigned short* w1b  = (unsigned short*)alloc((size_t)F_IN * HC * 2);
    unsigned short* w2b  = (unsigned short*)alloc((size_t)HC * OUTC * 2);
    float* h1    = (float*)alloc((size_t)NN * HC * 4);
    float* agg1  = (float*)alloc((size_t)NN * HC * 4);
    float* als1  = (float*)alloc((size_t)NN * HEADS * 4);
    float* ald1  = (float*)alloc((size_t)NN * HEADS * 4);
    float* m1    = (float*)alloc((size_t)NN * HEADS * 4);
    float* denom1= (float*)alloc((size_t)NN * HEADS * 4);
    float* e1    = (float*)alloc((size_t)EE * HEADS * 4);
    float* h2    = (float*)alloc((size_t)NN * OUTC * 4);
    float* als2  = (float*)alloc((size_t)NN * 4);
    float* ald2  = (float*)alloc((size_t)NN * 4);
    float* m2    = (float*)alloc((size_t)NN * 4);
    float* denom2= (float*)alloc((size_t)NN * 4);
    float* bnsum = (float*)alloc(HC * 4);
    float* bnsq  = (float*)alloc(HC * 4);
    float* scale = (float*)alloc(HC * 4);
    float* shift = (float*)alloc(HC * 4);
    // reuse: hn (bf16, 25.6MB) aliases h1 (51.2MB, dead after layer-1 aggregation);
    //        e2 aliases e1 (dead after layer 1)
    unsigned short* hn = (unsigned short*)h1;
    float* e2 = e1;

    const int B = 256;
    auto cdiv = [](int a, int b) { return (a + b - 1) / b; };

    // ---- bf16 conversions (packed pairs) ----
    k_cvt_bf16x2<<<cdiv(NN * F_IN / 2, B), B, 0, stream>>>(x, (unsigned int*)xb, NN * F_IN / 2);
    k_cvt_bf16x2<<<cdiv(F_IN * HC / 2, B), B, 0, stream>>>(W1, (unsigned int*)w1b, F_IN * HC / 2);
    k_cvt_bf16x2<<<cdiv(HC * OUTC / 2, B), B, 0, stream>>>(W2, (unsigned int*)w2b, HC * OUTC / 2);

    // ---- GEMM1: h1 = x @ W1 (WMMA bf16, fp32 acc) ----
    k_gemm_bf16_wmma<F_IN, 64, 4, HC><<<dim3(NN / 16, HC / 64), 128, 0, stream>>>(xb, w1b, h1);

    // ---- layer-1 attention softmax over edges (+self-loops) ----
    k_attn_dots<<<cdiv(NN * HEADS, B), B, 0, stream>>>(h1, a_src1, a_dst1, als1, ald1,
                                                       NN * HEADS, HEADS, HID);
    k_init_m<<<cdiv(NN * HEADS, B), B, 0, stream>>>(als1, ald1, m1, NN * HEADS);
    k_edge_max<<<cdiv(EE * HEADS, B), B, 0, stream>>>(ei, als1, ald1, e1, m1, EE, HEADS);
    k_init_denom<<<cdiv(NN * HEADS, B), B, 0, stream>>>(als1, ald1, m1, denom1, NN * HEADS);
    k_edge_expsum<<<cdiv(EE * HEADS, B), B, 0, stream>>>(ei, e1, m1, denom1, EE, HEADS);
    k_agg_init<<<cdiv(NN * HC / 4, B), B, 0, stream>>>(h1, als1, ald1, m1, denom1, agg1,
                                                       NN * HC / 4, HID);
    k_edge_aggregate<<<cdiv(EE * HC / 4, B), B, 0, stream>>>(ei, e1, denom1, h1, agg1,
                                                             EE, HEADS, HID);

    // ---- batchnorm (b1 folded in) + ReLU + bf16 ----
    k_zero<<<cdiv(2 * HC, B), B, 0, stream>>>(bnsum, 2 * HC);   // bnsum & bnsq adjacent
    k_bn_stats<<<250, 256, 0, stream>>>(agg1, b1, bnsum, bnsq, NN);
    k_bn_final<<<1, 256, 0, stream>>>(bnsum, bnsq, gamma, beta, scale, shift);
    k_norm_relu_bf16<<<cdiv(NN * HC / 2, B), B, 0, stream>>>(agg1, b1, scale, shift,
                                                             (unsigned int*)hn, NN * HC / 2);

    // ---- GEMM2: h2 = hn @ W2 (WMMA bf16) ----
    k_gemm_bf16_wmma<HC, 16, 1, OUTC><<<dim3(NN / 16, 1), 32, 0, stream>>>(hn, w2b, h2);

    // ---- layer-2 attention (H=1, C=16), aggregate straight into d_out ----
    k_attn_dots<<<cdiv(NN, B), B, 0, stream>>>(h2, a_src2, a_dst2, als2, ald2, NN, 1, OUTC);
    k_init_m<<<cdiv(NN, B), B, 0, stream>>>(als2, ald2, m2, NN);
    k_edge_max<<<cdiv(EE, B), B, 0, stream>>>(ei, als2, ald2, e2, m2, EE, 1);
    k_init_denom<<<cdiv(NN, B), B, 0, stream>>>(als2, ald2, m2, denom2, NN);
    k_edge_expsum<<<cdiv(EE, B), B, 0, stream>>>(ei, e2, m2, denom2, EE, 1);
    k_agg_init<<<cdiv(NN * OUTC / 4, B), B, 0, stream>>>(h2, als2, ald2, m2, denom2, out,
                                                         NN * OUTC / 4, OUTC);
    k_edge_aggregate<<<cdiv(EE * OUTC / 4, B), B, 0, stream>>>(ei, e2, denom2, h2, out,
                                                               EE, 1, OUTC);
    k_bias_out<<<cdiv(NN * OUTC, B), B, 0, stream>>>(out, b2, NN * OUTC);
}
